// GGCN_80925773791738
// MI455X (gfx1250) — matrix operand order repaired and checked by reading, not compiled, not run on hardware
//
#include <hip/hip_runtime.h>

// Problem constants from the reference
#define L_NODES 100000
#define NFEAT   256
#define JDIM    128
#define SSAMP   8

typedef __attribute__((ext_vector_type(16))) __bf16 v16bf;
typedef __attribute__((ext_vector_type(8)))  __bf16 v8bf;
typedef __attribute__((ext_vector_type(8)))  float  v8f;

// ---------------------------------------------------------------------------
// gfx1250 async copy: global -> LDS, 16 bytes per lane, tracked by ASYNCcnt.
// (Builtins for these are not probe-confirmed; use inline CDNA5 asm.)
// ---------------------------------------------------------------------------
__device__ __forceinline__ void async_ld16(unsigned lds_byte_off, const void* gaddr) {
  asm volatile("global_load_async_to_lds_b128 %0, %1, off"
               :: "v"(lds_byte_off), "v"(gaddr)
               : "memory");
}
__device__ __forceinline__ void wait_async0() {
  asm volatile("s_wait_asynccnt 0x0" ::: "memory");
}
__device__ __forceinline__ void wait_async2() {
  asm volatile("s_wait_asynccnt 0x2" ::: "memory");
}
// Generic pointer to LDS -> 32-bit LDS byte offset (low 32 bits of flat addr).
__device__ __forceinline__ unsigned lds_off(const void* p) {
  return (unsigned)(__SIZE_TYPE__)p;
}

// ---------------------------------------------------------------------------
// Pack a (256 x 128) row-major f32 weight matrix into WMMA 16-bit B-fragment
// order (bf16).  For V_WMMA_F32_16X16X32_BF16 the B matrix (32x16) layout is:
//   lane = (kHalf<<4) | n ; element e (0..15) holds B[(e&8)*2 + kHalf*8 + (e&7)][n]
// Packed index: (((kc*8 + nt)*32 + lane)*16 + e), 16 bf16 = 32 bytes per lane.
// ---------------------------------------------------------------------------
__global__ void k_pack(const float* __restrict__ W, __bf16* __restrict__ P) {
  int t = blockIdx.x * 256 + threadIdx.x;        // 32768 total elements
  int e    = t & 15;
  int lane = (t >> 4) & 31;
  int nt   = (t >> 9) & 7;
  int kc   = t >> 12;
  int K    = kc * 32 + (e & 8) * 2 + (lane >> 4) * 8 + (e & 7);     // 0..255
  int col  = nt * 16 + (lane & 15);              // 0..127
  P[t] = (__bf16)W[K * JDIM + col];
}

// A-fragment (16x32, 16-bit): lane (m = lane&15, half = lane>>4):
// elements 0..7 = K half*8+0..7, elements 8..15 = K 16+half*8+0..7.
__device__ __forceinline__ v16bf a_frag_f32(const float* p) {
  v16bf a;
#pragma unroll
  for (int i = 0; i < 8; ++i) {
    a[i]     = (__bf16)p[i];
    a[i + 8] = (__bf16)p[i + 16];
  }
  return a;
}
__device__ __forceinline__ v16bf a_frag_bf16(const __bf16* p) {
  v8bf lo = *(const v8bf*)(p);
  v8bf hi = *(const v8bf*)(p + 16);
  return __builtin_shufflevector(lo, hi, 0, 1, 2, 3, 4, 5, 6, 7,
                                 8, 9, 10, 11, 12, 13, 14, 15);
}

// LDS layout constants
#define SW_BYTES   65536            // packed weight frags (2048 v16bf)
#define AROW_PITCH 272              // 256B bf16 row padded to 17*16B (no bank conflicts)
#define ABUF_BYTES (32 * AROW_PITCH)   // 8704: 32 gathered rows per sample
#define XROW_PITCH 1040             // 1024B f32 row padded to 65*16B
#define XBUF_BYTES (16 * XROW_PITCH)   // 16640

// ---------------------------------------------------------------------------
// K1: H = relu(X @ W_h1 + b_h1), stored bf16 (L x 128).
// Weights + 16 X rows async-staged to LDS; 8 waves = 8 N-tiles per 16-row block.
// ---------------------------------------------------------------------------
__global__ void k_h(const float* __restrict__ X,
                    const v16bf* __restrict__ Wp,
                    const float* __restrict__ bh,
                    __bf16* __restrict__ Hb) {
  extern __shared__ char smem[];                 // SW_BYTES + XBUF_BYTES
  v16bf* sW = (v16bf*)smem;
  const unsigned base = lds_off(smem);
  const int tid = threadIdx.x;

  // stage packed W_h1 fragments (64 KB = 4096 x 16B chunks)
#pragma unroll
  for (int i = 0; i < 16; ++i) {
    const int c = tid + i * 256;
    async_ld16(base + c * 16, (const char*)Wp + c * 16);
  }
  // stage 16 rows of X (1 KB each -> 1024 x 16B chunks, padded pitch)
  const long rowBase = (long)blockIdx.x * 16;
#pragma unroll
  for (int i = 0; i < 4; ++i) {
    const int c  = tid + i * 256;                // 0..1023
    const int r  = c >> 6;                       // 0..15
    const int ck = c & 63;                       // 16B chunk within row
    async_ld16(base + SW_BYTES + r * XROW_PITCH + ck * 16,
               (const char*)(X + (rowBase + r) * NFEAT) + ck * 16);
  }
  wait_async0();
  __syncthreads();

  const int w = tid >> 5, lane = tid & 31;
  const int m = lane & 15, half = lane >> 4;
  const float* xrow = (const float*)(smem + SW_BYTES + m * XROW_PITCH);

  v8f c = {};
#pragma unroll
  for (int kc = 0; kc < 8; ++kc) {               // K = 256 in chunks of 32
    v16bf a = a_frag_f32(xrow + kc * 32 + half * 8);
    v16bf b = sW[(kc * 8 + w) * 32 + lane];
    c = __builtin_amdgcn_wmma_f32_16x16x32_bf16(false, a, false, b,
                                                (short)0, c, false, false);
  }
  const float bias = bh[w * 16 + m];
  const long orow = (long)blockIdx.x * 16 + half * 8;  // C layout: row = r + 8*half
#pragma unroll
  for (int r = 0; r < 8; ++r) {
    float v = c[r] + bias;
    v = v > 0.f ? v : 0.f;
    Hb[(orow + r) * JDIM + w * 16 + m] = (__bf16)v;
  }
}

// ---------------------------------------------------------------------------
// K2 (dominant): E = relu(mean_s relu(cat(H[idx0_s], H[idx1_s]) @ W_g1 + b)).
// bf16 H (25.6 MB) is L2-resident; gathered rows are async-staged into a
// double-buffered LDS tile so the 8 waves share one gather and the sample-s+1
// gather latency hides under the sample-s WMMAs.
// ---------------------------------------------------------------------------
__global__ void k_pair(const __bf16* __restrict__ Hb,
                       const v16bf* __restrict__ Wp,
                       const float* __restrict__ bg,
                       const int* __restrict__ idx0,
                       const int* __restrict__ idx1,
                       __bf16* __restrict__ Eb) {
  extern __shared__ char smem[];                 // SW_BYTES + 2*ABUF_BYTES
  v16bf* sW = (v16bf*)smem;
  const unsigned base = lds_off(smem);
  const int tid = threadIdx.x;

  // stage packed W_g1 fragments (64 KB) asynchronously
#pragma unroll
  for (int i = 0; i < 16; ++i) {
    const int c = tid + i * 256;
    async_ld16(base + c * 16, (const char*)Wp + c * 16);
  }

  const int w = tid >> 5, lane = tid & 31;
  const int m = lane & 15, half = lane >> 4;

  // Gather stage: 32 rows (16 from idx0, 16 from idx1) x 256B, padded pitch.
  // 512 x 16B chunks -> 2 async loads per thread.
  auto issue_gather = [&](int s, int buf) {
    const unsigned boff = base + SW_BYTES + (unsigned)buf * ABUF_BYTES;
#pragma unroll
    for (int i = 0; i < 2; ++i) {
      const int c  = tid + i * 256;              // 0..511
      const int r  = c >> 4;                     // 0..31
      const int ck = c & 15;                     // 16B chunk within row
      const int node = blockIdx.x * 16 + (r & 15);
      const int src  = (r < 16) ? idx0[s * L_NODES + node]
                                : idx1[s * L_NODES + node];
      async_ld16(boff + (unsigned)(r * AROW_PITCH + ck * 16),
                 (const char*)(Hb + (long)src * JDIM) + ck * 16);
    }
  };

  issue_gather(0, 0);

  v8f meanAcc = {};
  const float bias = bg[w * 16 + m];

  for (int s = 0; s < SSAMP; ++s) {
    const int cur = s & 1;
    if (s + 1 < SSAMP) { issue_gather(s + 1, 1 - cur); wait_async2(); }
    else               { wait_async0(); }
    __syncthreads();                             // weights + buf[cur] visible

    const char* aBuf = smem + SW_BYTES + cur * ABUF_BYTES;
    v8f c = {};
#pragma unroll
    for (int kc = 0; kc < 8; ++kc) {             // K = 2J = 256
      const int r = (kc < 4) ? m : (16 + m);     // idx0 rows then idx1 rows
      const __bf16* arow =
          (const __bf16*)(aBuf + r * AROW_PITCH + ((kc & 3) * 32 + half * 8) * 2);
      v16bf a = a_frag_bf16(arow);
      v16bf b = sW[(kc * 8 + w) * 32 + lane];
      c = __builtin_amdgcn_wmma_f32_16x16x32_bf16(false, a, false, b,
                                                  (short)0, c, false, false);
    }
#pragma unroll
    for (int r = 0; r < 8; ++r) {
      float v = c[r] + bias;
      meanAcc[r] += v > 0.f ? v : 0.f;
    }
    __syncthreads();                             // readers done before buf reuse
  }

  const long orow = (long)blockIdx.x * 16 + half * 8;
#pragma unroll
  for (int r = 0; r < 8; ++r) {
    float v = meanAcc[r] * (1.0f / SSAMP);
    v = v > 0.f ? v : 0.f;                       // relu of mean of relus
    Eb[(orow + r) * JDIM + w * 16 + m] = (__bf16)v;
  }
}

// ---------------------------------------------------------------------------
// K3: E2 = relu(cat(H, E) @ W_g1 + b_g1); own-row A tile async-staged once.
// ---------------------------------------------------------------------------
__global__ void k_final(const __bf16* __restrict__ Hb,
                        const __bf16* __restrict__ Eb,
                        const v16bf* __restrict__ Wp,
                        const float* __restrict__ bg,
                        __bf16* __restrict__ E2b) {
  extern __shared__ char smem[];                 // SW_BYTES + ABUF_BYTES
  v16bf* sW = (v16bf*)smem;
  const unsigned base = lds_off(smem);
  const int tid = threadIdx.x;

#pragma unroll
  for (int i = 0; i < 16; ++i) {
    const int c = tid + i * 256;
    async_ld16(base + c * 16, (const char*)Wp + c * 16);
  }
  // stage 16 rows of H then 16 rows of E for this block (512 x 16B chunks)
#pragma unroll
  for (int i = 0; i < 2; ++i) {
    const int c  = tid + i * 256;
    const int r  = c >> 4;
    const int ck = c & 15;
    const long node = (long)blockIdx.x * 16 + (r & 15);
    const __bf16* srcRow = ((r < 16) ? Hb : Eb) + node * JDIM;
    async_ld16(base + SW_BYTES + (unsigned)(r * AROW_PITCH + ck * 16),
               (const char*)srcRow + ck * 16);
  }
  wait_async0();
  __syncthreads();

  const int w = tid >> 5, lane = tid & 31;
  const int m = lane & 15, half = lane >> 4;

  v8f c = {};
#pragma unroll
  for (int kc = 0; kc < 8; ++kc) {
    const int r = (kc < 4) ? m : (16 + m);       // H rows then E rows
    const __bf16* arow =
        (const __bf16*)(smem + SW_BYTES + r * AROW_PITCH +
                        ((kc & 3) * 32 + half * 8) * 2);
    v16bf a = a_frag_bf16(arow);
    v16bf b = sW[(kc * 8 + w) * 32 + lane];
    c = __builtin_amdgcn_wmma_f32_16x16x32_bf16(false, a, false, b,
                                                (short)0, c, false, false);
  }
  const float bias = bg[w * 16 + m];
  const long orow = (long)blockIdx.x * 16 + half * 8;
#pragma unroll
  for (int r = 0; r < 8; ++r) {
    float v = c[r] + bias;
    v = v > 0.f ? v : 0.f;
    E2b[(orow + r) * JDIM + w * 16 + m] = (__bf16)v;
  }
}

// ---------------------------------------------------------------------------
// K4: out = E2 @ W_f + b_f   (L x 2) — trivially memory bound.
// ---------------------------------------------------------------------------
__global__ void k_out(const __bf16* __restrict__ E2,
                      const float* __restrict__ Wf,
                      const float* __restrict__ bf_,
                      float* __restrict__ out) {
  long i = (long)blockIdx.x * 256 + threadIdx.x;
  if (i >= L_NODES) return;
  float a0 = bf_[0], a1 = bf_[1];
  const __bf16* e = E2 + i * JDIM;
#pragma unroll 4
  for (int j = 0; j < JDIM; ++j) {
    float v = (float)e[j];
    a0 += v * Wf[2 * j];
    a1 += v * Wf[2 * j + 1];
  }
  out[2 * i]     = a0;
  out[2 * i + 1] = a1;
}

// ---------------------------------------------------------------------------
extern "C" void kernel_launch(void* const* d_in, const int* in_sizes, int n_in,
                              void* d_out, int out_size, void* d_ws, size_t ws_size,
                              hipStream_t stream) {
  const float* X    = (const float*)d_in[0];
  const float* Wh1  = (const float*)d_in[1];
  const float* bh1  = (const float*)d_in[2];
  const float* Wg1  = (const float*)d_in[3];
  const float* bg1  = (const float*)d_in[4];
  const float* Wf   = (const float*)d_in[5];
  const float* bf_  = (const float*)d_in[6];
  const int*   idx0 = (const int*)d_in[7];
  const int*   idx1 = (const int*)d_in[8];

  char* ws = (char*)d_ws;
  __bf16* Hb  = (__bf16*)(ws + 0);            // 25.6 MB  (L x 128 bf16)
  __bf16* Eb  = (__bf16*)(ws + 25600000);     // 25.6 MB
  __bf16* E2b = (__bf16*)(ws + 51200000);     // 25.6 MB
  __bf16* Ph  = (__bf16*)(ws + 76800000);     // 64 KB packed W_h1 frags
  __bf16* Pg  = (__bf16*)(ws + 76865536);     // 64 KB packed W_g1 frags

  const int ROWBLOCKS = L_NODES / 16;         // 6250, exact

  k_pack<<<128, 256, 0, stream>>>(Wh1, Ph);
  k_pack<<<128, 256, 0, stream>>>(Wg1, Pg);
  k_h    <<<ROWBLOCKS, 256, SW_BYTES + XBUF_BYTES,     stream>>>(X, (const v16bf*)Ph, bh1, Hb);
  k_pair <<<ROWBLOCKS, 256, SW_BYTES + 2 * ABUF_BYTES, stream>>>(Hb, (const v16bf*)Pg, bg1, idx0, idx1, Eb);
  k_final<<<ROWBLOCKS, 256, SW_BYTES + ABUF_BYTES,     stream>>>(Hb, Eb, (const v16bf*)Pg, bg1, E2b);
  k_out  <<<(L_NODES + 255) / 256, 256, 0, stream>>>(E2b, Wf, bf_, (float*)d_out);
}
